// MultiHeadAttention_49821620633656
// MI455X (gfx1250) — matrix operand order repaired and checked by reading, not compiled, not run on hardware
//
#include <hip/hip_runtime.h>
#include <hip/hip_bf16.h>

#define BATCH 2
#define SEQ   2048
#define EMB   1024
#define HEADS 16
#define HDIM  64

typedef __attribute__((ext_vector_type(16))) __bf16 bf16x16;
typedef __attribute__((ext_vector_type(8)))  __bf16 bf16x8;
typedef __attribute__((ext_vector_type(8)))  float  f32x8;

#define WMMA_BF16(a, b, c) \
  __builtin_amdgcn_wmma_f32_16x16x32_bf16(false, (a), false, (b), (short)0, (c), false, false)

static __device__ __forceinline__ __bf16 f2bf(float f) {
  unsigned u = __builtin_bit_cast(unsigned, f);
  u += 0x7FFFu + ((u >> 16) & 1u);          // round to nearest even
  return __builtin_bit_cast(__bf16, (unsigned short)(u >> 16));
}

// A fragment: 16x32 bf16, row-major source with leading dim `ld`.
// Lane L (L<16): row=L, K chunk base 0; lane L>=16: row=L-16, K chunk base 8.
// elems 0..7 -> K = base+0..7 ; elems 8..15 -> K = 16+base+0..7  (ISA 7.12.2)
static __device__ __forceinline__ bf16x16 load_a_frag(const __bf16* base, int ld, int lane) {
  const int row = lane & 15;
  const int off = (lane < 16) ? 0 : 8;
  const __bf16* p = base + (size_t)row * ld + off;
  bf16x8 lo = *reinterpret_cast<const bf16x8*>(p);
  bf16x8 hi = *reinterpret_cast<const bf16x8*>(p + 16);
  bf16x16 r;
#pragma unroll
  for (int i = 0; i < 8; ++i) { r[i] = lo[i]; r[i + 8] = hi[i]; }
  return r;
}

// B fragment: 32x16 bf16. Source is "N-major": 16 rows (one per N) each
// contiguous in K with leading dim `ld`. Lane L: col N=L&15, K chunk base 0 (L<16)
// or 8 (L>=16); elems 0..7 -> K = base+0..7, elems 8..15 -> K = 16+base+0..7.
static __device__ __forceinline__ bf16x16 load_b_frag(const __bf16* base, int ld, int lane) {
  const int n = lane & 15;
  const int off = (lane < 16) ? 0 : 8;
  const __bf16* p = base + (size_t)n * ld + off;
  bf16x8 lo = *reinterpret_cast<const bf16x8*>(p);
  bf16x8 hi = *reinterpret_cast<const bf16x8*>(p + 16);
  bf16x16 r;
#pragma unroll
  for (int i = 0; i < 8; ++i) { r[i] = lo[i]; r[i + 8] = hi[i]; }
  return r;
}

__global__ __launch_bounds__(256) void cvt_bf16(const float* __restrict__ src,
                                                __bf16* __restrict__ dst, int n) {
  int i = blockIdx.x * blockDim.x + threadIdx.x;
  if (i < n) dst[i] = f2bf(src[i]);
}

// qkv = x @ w_qkv^T  (per batch: M=SEQ, N=3*EMB, K=EMB).
// Register-blocked: each wave computes a 64x32 output block (4 M-tiles x 2
// N-tiles) -> 8 WMMAs per K-step fed by 12 b128 loads.
// Scatter to Q [B,H,S,D], K [B,H,S,D], V^T [B,H,D,S] as bf16.
__global__ __launch_bounds__(256) void qkv_gemm(const __bf16* __restrict__ xbf,
                                                const __bf16* __restrict__ wbf,
                                                __bf16* __restrict__ qm,
                                                __bf16* __restrict__ km,
                                                __bf16* __restrict__ vtm) {
  const int lane = threadIdx.x & 31;
  // Wave-uniform tile index forced into SGPRs: scalar addressing + scalar
  // branches in the scatter epilogue (the raw threadIdx-derived value is
  // formally divergent to the compiler).
  const int t = __builtin_amdgcn_readfirstlane(blockIdx.x * 8 + (threadIdx.x >> 5));
  const int b = blockIdx.z;
  const int n0 = (t % 96) * 32;                 // (3*EMB)/32 = 96 N-blocks
  const int m0 = (t / 96) * 64;                 // SEQ/64 = 32 M-blocks
  const __bf16* abase = xbf + ((size_t)b * SEQ + m0) * EMB;
  const __bf16* bbase = wbf + (size_t)n0 * EMB; // w_qkv row-major [f,e] == N-major

  f32x8 acc[4][2];
#pragma unroll
  for (int i = 0; i < 4; ++i)
#pragma unroll
    for (int j = 0; j < 2; ++j) acc[i][j] = {};

  for (int k0 = 0; k0 < EMB; k0 += 32) {
    bf16x16 a[4], bb[2];
#pragma unroll
    for (int i = 0; i < 4; ++i)
      a[i] = load_a_frag(abase + (size_t)(i * 16) * EMB + k0, EMB, lane);
#pragma unroll
    for (int j = 0; j < 2; ++j)
      bb[j] = load_b_frag(bbase + (size_t)(j * 16) * EMB + k0, EMB, lane);
#pragma unroll
    for (int i = 0; i < 4; ++i)
#pragma unroll
      for (int j = 0; j < 2; ++j)
        acc[i][j] = WMMA_BF16(a[i], bb[j], acc[i][j]);
  }

  const int lane15 = lane & 15;
  const int hi8 = (lane < 16) ? 0 : 8;
#pragma unroll
  for (int j = 0; j < 2; ++j) {
    // A 16-aligned column tile never crosses a 1024 (which) or 64 (head)
    // boundary, so decode from the scalar tile base only.
    const int fbase = n0 + j * 16;      // SGPR
    const int which = fbase >> 10;      // SGPR -> scalar branch
    const int h     = (fbase >> 6) & 15;
    const int dbase = fbase & 63;
    const int d = dbase + lane15;       // per-lane
    if (which == 2) {
      // V^T: [b,h,d,s], s contiguous per row
      __bf16* dst = vtm + (((size_t)b * HEADS + h) * HDIM + d) * SEQ + m0;
#pragma unroll
      for (int i = 0; i < 4; ++i)
#pragma unroll
        for (int r = 0; r < 8; ++r)
          dst[i * 16 + r + hi8] = f2bf(acc[i][j][r]);
    } else {
      __bf16* base = (which == 0) ? qm : km;
      __bf16* dst = base + (((size_t)b * HEADS + h) * SEQ + m0) * HDIM + d;
#pragma unroll
      for (int i = 0; i < 4; ++i)
#pragma unroll
        for (int r = 0; r < 8; ++r)
          dst[(size_t)(i * 16 + r + hi8) * HDIM] = f2bf(acc[i][j][r]);
    }
  }
}

// Causal flash attention. One wave per 16-row query tile; key chunks of 32.
// ctx written as [B, S, H*D] bf16 so the output GEMM reads it row-major.
__global__ __launch_bounds__(128) void flash_attn(const __bf16* __restrict__ qm,
                                                  const __bf16* __restrict__ km,
                                                  const __bf16* __restrict__ vtm,
                                                  __bf16* __restrict__ ctx) {
  __shared__ __bf16 pbuf[4][16 * 32];           // per-wave 16x32 P tile
  const int lane = threadIdx.x & 31;
  const int wave = __builtin_amdgcn_readfirstlane(threadIdx.x >> 5);
  const int bh = blockIdx.x;                    // b*HEADS + h
  const int b = bh >> 4, h = bh & 15;
  const int q0 = (blockIdx.y * 4 + wave) * 16;  // SGPR
  const __bf16* qb = qm + ((size_t)bh * SEQ + q0) * HDIM;
  const __bf16* kb = km + (size_t)bh * SEQ * HDIM;
  const __bf16* vb = vtm + (size_t)bh * HDIM * SEQ;
  const int lane15 = lane & 15;
  const int hi8 = (lane < 16) ? 0 : 8;

  bf16x16 aq0 = load_a_frag(qb, HDIM, lane);        // d = 0..31
  bf16x16 aq1 = load_a_frag(qb + 32, HDIM, lane);   // d = 32..63

  f32x8 cacc0 = {}, cacc1 = {}, cacc2 = {}, cacc3 = {};
  float mrow[8], lrow[8];
#pragma unroll
  for (int r = 0; r < 8; ++r) { mrow[r] = -1e30f; lrow[r] = 0.0f; }

  __bf16* pw = &pbuf[wave][0];

  for (int kc = 0; kc < q0 + 16; kc += 32) {
    f32x8 s0 = {}, s1 = {};
    {
      bf16x16 b0 = load_b_frag(kb + (size_t)kc * HDIM, HDIM, lane);
      bf16x16 b1 = load_b_frag(kb + (size_t)kc * HDIM + 32, HDIM, lane);
      s0 = WMMA_BF16(aq0, b0, s0);
      s0 = WMMA_BF16(aq1, b1, s0);
    }
    {
      bf16x16 b0 = load_b_frag(kb + (size_t)(kc + 16) * HDIM, HDIM, lane);
      bf16x16 b1 = load_b_frag(kb + (size_t)(kc + 16) * HDIM + 32, HDIM, lane);
      s1 = WMMA_BF16(aq0, b0, s1);
      s1 = WMMA_BF16(aq1, b1, s1);
    }
    const int col0 = kc + lane15;
    const int col1 = col0 + 16;
#pragma unroll
    for (int r = 0; r < 8; ++r) {
      const int row = q0 + r + hi8;
      float v0 = (col0 <= row) ? s0[r] * 0.125f : -1e30f;   // 1/sqrt(64)
      float v1 = (col1 <= row) ? s1[r] * 0.125f : -1e30f;
      float mx = fmaxf(v0, v1);
      for (int o = 1; o < 16; o <<= 1) mx = fmaxf(mx, __shfl_xor(mx, o, 32));
      const float mnew  = fmaxf(mrow[r], mx);
      const float alpha = __expf(mrow[r] - mnew);
      const float p0 = __expf(v0 - mnew);
      const float p1 = __expf(v1 - mnew);
      float rs = p0 + p1;
      for (int o = 1; o < 16; o <<= 1) rs += __shfl_xor(rs, o, 32);
      lrow[r] = lrow[r] * alpha + rs;
      mrow[r] = mnew;
      cacc0[r] *= alpha; cacc1[r] *= alpha; cacc2[r] *= alpha; cacc3[r] *= alpha;
      pw[(r + hi8) * 32 + lane15]      = f2bf(p0);
      pw[(r + hi8) * 32 + 16 + lane15] = f2bf(p1);
    }
    // Re-read P (16x32) from per-wave LDS in A-fragment layout.
    // Same-wave DS ops are in-order; no cross-wave sharing -> no block barrier.
    bf16x16 ap = load_a_frag(pw, 32, lane);
    cacc0 = WMMA_BF16(ap, load_b_frag(vb + kc,                    SEQ, lane), cacc0);
    cacc1 = WMMA_BF16(ap, load_b_frag(vb + (size_t)16 * SEQ + kc, SEQ, lane), cacc1);
    cacc2 = WMMA_BF16(ap, load_b_frag(vb + (size_t)32 * SEQ + kc, SEQ, lane), cacc2);
    cacc3 = WMMA_BF16(ap, load_b_frag(vb + (size_t)48 * SEQ + kc, SEQ, lane), cacc3);
  }
  __bf16* cdst = ctx + ((size_t)b * SEQ + q0) * EMB + (size_t)h * HDIM + lane15;
#pragma unroll
  for (int r = 0; r < 8; ++r) {
    const float inv = 1.0f / lrow[r];
    __bf16* p = cdst + (size_t)(r + hi8) * EMB;
    p[0]  = f2bf(cacc0[r] * inv);
    p[16] = f2bf(cacc1[r] * inv);
    p[32] = f2bf(cacc2[r] * inv);
    p[48] = f2bf(cacc3[r] * inv);
  }
}

// out = ctx @ w_out^T + b_out   (M = B*SEQ flattened, N = EMB, K = EMB).
// Register-blocked 64x32 per wave, fp32 out.
__global__ __launch_bounds__(256) void out_gemm(const __bf16* __restrict__ ctx,
                                                const __bf16* __restrict__ wbf,
                                                const float* __restrict__ bias,
                                                float* __restrict__ out) {
  const int lane = threadIdx.x & 31;
  const int t = __builtin_amdgcn_readfirstlane(blockIdx.x * 8 + (threadIdx.x >> 5));
  const int n0 = (t & 31) * 32;                 // EMB/32 = 32 N-blocks
  const int m0 = (t >> 5) * 64;                 // (B*SEQ)/64 = 64 M-blocks
  const __bf16* abase = ctx + (size_t)m0 * EMB;
  const __bf16* bbase = wbf + (size_t)n0 * EMB;

  f32x8 acc[4][2];
#pragma unroll
  for (int i = 0; i < 4; ++i)
#pragma unroll
    for (int j = 0; j < 2; ++j) acc[i][j] = {};

  for (int k0 = 0; k0 < EMB; k0 += 32) {
    bf16x16 a[4], bb[2];
#pragma unroll
    for (int i = 0; i < 4; ++i)
      a[i] = load_a_frag(abase + (size_t)(i * 16) * EMB + k0, EMB, lane);
#pragma unroll
    for (int j = 0; j < 2; ++j)
      bb[j] = load_b_frag(bbase + (size_t)(j * 16) * EMB + k0, EMB, lane);
#pragma unroll
    for (int i = 0; i < 4; ++i)
#pragma unroll
      for (int j = 0; j < 2; ++j)
        acc[i][j] = WMMA_BF16(a[i], bb[j], acc[i][j]);
  }

  const int lane15 = lane & 15;
  const int hi8 = (lane < 16) ? 0 : 8;
#pragma unroll
  for (int j = 0; j < 2; ++j) {
    const int f = n0 + j * 16 + lane15;
    const float bv = bias[f];
    float* dst = out + (size_t)m0 * EMB + f;
#pragma unroll
    for (int i = 0; i < 4; ++i)
#pragma unroll
      for (int r = 0; r < 8; ++r)
        dst[(size_t)(i * 16 + r + hi8) * EMB] = acc[i][j][r] + bv;
  }
}

extern "C" void kernel_launch(void* const* d_in, const int* in_sizes, int n_in,
                              void* d_out, int out_size, void* d_ws, size_t ws_size,
                              hipStream_t stream) {
  const float* x     = (const float*)d_in[0];
  const float* w_qkv = (const float*)d_in[1];
  const float* w_out = (const float*)d_in[2];
  const float* b_out = (const float*)d_in[3];
  float* out = (float*)d_out;

  const size_t NX   = (size_t)BATCH * SEQ * EMB;          // 4M
  const size_t NWQ  = (size_t)3 * EMB * EMB;              // 3M
  const size_t NWO  = (size_t)EMB * EMB;                  // 1M
  const size_t NQKV = (size_t)BATCH * HEADS * SEQ * HDIM; // 4M

  __bf16* ws   = (__bf16*)d_ws;                           // 48 MiB total in bf16
  __bf16* xbf  = ws;
  __bf16* wqbf = xbf + NX;
  __bf16* wobf = wqbf + NWQ;
  __bf16* qm   = wobf + NWO;
  __bf16* km   = qm + NQKV;
  __bf16* vtm  = km + NQKV;
  __bf16* ctx  = vtm + NQKV;

  cvt_bf16<<<(int)(NX / 256),  256, 0, stream>>>(x, xbf, (int)NX);
  cvt_bf16<<<(int)(NWQ / 256), 256, 0, stream>>>(w_qkv, wqbf, (int)NWQ);
  cvt_bf16<<<(int)(NWO / 256), 256, 0, stream>>>(w_out, wobf, (int)NWO);

  qkv_gemm<<<dim3(384, 1, BATCH), 256, 0, stream>>>(xbf, wqbf, qm, km, vtm);
  flash_attn<<<dim3(BATCH * HEADS, SEQ / 64), 128, 0, stream>>>(qm, km, vtm, ctx);
  out_gemm<<<256, 256, 0, stream>>>(ctx, wobf, b_out, out);
}
